// BiMambaEncoder_54185307406706
// MI455X (gfx1250) — compile-verified, matrix operand order checked by reading, not run on hardware
//
#include <hip/hip_runtime.h>
#include <hip/hip_bf16.h>
#include <math.h>
#include <stdint.h>

// ---------------- problem constants ----------------
#define BATCH  2
#define SEQ    1024
#define DM     512
#define EDIM   1024          // EXPAND * D_MODEL
#define NST    16            // state size N
#define DTR    32            // DT_RANK
#define DFF    1024
#define DCONVK 4
#define MROWS  (BATCH*SEQ)   // 2048 token rows
#define EPS_RMS 1e-5f

typedef __bf16 bf16_t;
typedef __attribute__((ext_vector_type(16))) __bf16 v16bf;
typedef __attribute__((ext_vector_type(8)))  __bf16 v8bf;
typedef __attribute__((ext_vector_type(8)))  float  v8f;

// ---------------- elementwise helpers ----------------
__global__ void k_reverse(const float* __restrict__ x, float* __restrict__ xr) {
    int idx = blockIdx.x * blockDim.x + threadIdx.x;
    if (idx >= MROWS * DM) return;
    int d = idx % DM, row = idx / DM;
    int b = row / SEQ, t = row % SEQ;
    xr[(size_t)(b * SEQ + t) * DM + d] = x[(size_t)(b * SEQ + (SEQ - 1 - t)) * DM + d];
}

// fp32 (K x N) row-major  ->  bf16 (N x K) row-major (transposed for WMMA B operand)
__global__ void k_wt(const float* __restrict__ W, bf16_t* __restrict__ Wt, int K, int Nn) {
    int idx = blockIdx.x * blockDim.x + threadIdx.x;
    if (idx >= K * Nn) return;
    int k = idx / Nn, n = idx % Nn;
    Wt[(size_t)n * K + k] = (bf16_t)W[idx];
}

// RMSNorm over last dim D (one block per row)
__global__ void k_rmsnorm(const float* __restrict__ X, const float* __restrict__ w,
                          float* __restrict__ Yf, bf16_t* __restrict__ Yb, int D) {
    int row = blockIdx.x, tid = threadIdx.x;
    __shared__ float red[256];
    const float* xr = X + (size_t)row * D;
    float s = 0.f;
    for (int i = tid; i < D; i += 256) { float v = xr[i]; s += v * v; }
    red[tid] = s; __syncthreads();
    for (int o = 128; o > 0; o >>= 1) { if (tid < o) red[tid] += red[tid + o]; __syncthreads(); }
    float scale = rsqrtf(red[0] / (float)D + EPS_RMS);
    for (int i = tid; i < D; i += 256) {
        float v = xr[i] * scale * w[i];
        if (Yf) Yf[(size_t)row * D + i] = v;
        if (Yb) Yb[(size_t)row * D + i] = (bf16_t)v;
    }
}

// depthwise causal conv (DCONV=4) + bias + SiLU on xh = xz[:, :EDIM]
__global__ void k_conv(const float* __restrict__ xz, const float* __restrict__ cw,
                       const float* __restrict__ cb, float* __restrict__ xc,
                       bf16_t* __restrict__ xcb) {
    int idx = blockIdx.x * blockDim.x + threadIdx.x;
    if (idx >= MROWS * EDIM) return;
    int e = idx % EDIM, row = idx / EDIM;
    int b = row / SEQ, t = row % SEQ;
    float acc = cb[e];
#pragma unroll
    for (int k = 0; k < DCONVK; k++) {
        int tt = t - (DCONVK - 1) + k;
        if (tt >= 0)
            acc += xz[(size_t)(b * SEQ + tt) * (2 * EDIM) + e] * cw[e * DCONVK + k];
    }
    float sv = acc / (1.f + __expf(-acc));   // SiLU
    xc[idx] = sv;
    xcb[idx] = (bf16_t)sv;
}

// ---------------- LDS-tiled, async double-buffered bf16 WMMA GEMM ----------------
// C[M,N] = A[M,K](bf16,row,lda) * B[K,N], B pre-transposed Bt[N,K](bf16,row).
// Block = 256 thr (8 waves), block tile 128 x BN (BN = 128 or 64).
// Waves in 4x2 grid: wave tile 32 x BN/2 -> 2 A-frags x (BN/32) B-frags = 2*BN/32 WMMAs/chunk.
// K staged in 32-wide chunks via global_load_async_to_lds_b128 (ASYNCcnt), double buffered.
// LDS rows padded to 80B: 16B alignment for ds_load_b128 + 20-bank stagger per row.
template<int BN, int ACT, bool BIAS, bool RES>
__global__ void __launch_bounds__(256)
k_gemm(const bf16_t* __restrict__ A, int lda,
       const bf16_t* __restrict__ Bt,
       const float* __restrict__ bias,
       const float* __restrict__ res,
       float* __restrict__ Cf, bf16_t* __restrict__ Cb,
       int Mr, int Nn, int K) {
    constexpr int NF     = BN / 32;           // B fragments per wave (4 or 2)
    constexpr int LDSA_B = 128 * 80;          // A: 128 rows * 80B
    constexpr int LDSB_B = BN * 80;           // B: BN rows * 80B
    constexpr int BUF_B  = LDSA_B + LDSB_B;
    __shared__ __align__(16) char smem[2 * BUF_B];

    const int t    = threadIdx.x;
    const int wv   = t >> 5;
    const int lane = t & 31;
    const int hf   = lane >> 4;               // half-wave -> K sub-chunk
    const int lm   = lane & 15;
    const int wm   = wv & 3;                  // 4 M strips of 32 rows
    const int wn   = wv >> 2;                 // 2 N strips of BN/2 cols
    const int nb   = Nn / BN;
    const int bm   = blockIdx.x / nb;
    const int bn   = blockIdx.x % nb;
    const int gm0  = bm * 128;
    const int gn0  = bn * BN;

    // staging: 16B chunks (8 bf16). A tile = 512 chunks (2/thread).
    // B tile = BN*4 chunks (2/thread for BN=128, 1/thread for BN=64).
    const int arA0 = t >> 2,         acA0 = t & 3;
    const int arA1 = (t + 256) >> 2, acA1 = (t + 256) & 3;
    const int brB0 = t >> 2,         bcB0 = t & 3;
    const int brB1 = (t + 256) >> 2, bcB1 = (t + 256) & 3;
    const unsigned smem0 = (unsigned)(uintptr_t)smem;   // LDS byte address (low 32 bits)

    auto stage = [&](int kc, int buf) {
        const int k0 = kc * 32;
        const unsigned sb = smem0 + buf * BUF_B;
        const bf16_t* ga0 = A  + (size_t)(gm0 + arA0) * lda + k0 + acA0 * 8;
        const bf16_t* ga1 = A  + (size_t)(gm0 + arA1) * lda + k0 + acA1 * 8;
        unsigned la0 = sb + arA0 * 80 + acA0 * 16;
        unsigned la1 = sb + arA1 * 80 + acA1 * 16;
        asm volatile("global_load_async_to_lds_b128 %0, %1, off" :: "v"(la0), "v"(ga0) : "memory");
        asm volatile("global_load_async_to_lds_b128 %0, %1, off" :: "v"(la1), "v"(ga1) : "memory");
        const bf16_t* gb0 = Bt + (size_t)(gn0 + brB0) * K + k0 + bcB0 * 8;
        unsigned lb0 = sb + LDSA_B + brB0 * 80 + bcB0 * 16;
        asm volatile("global_load_async_to_lds_b128 %0, %1, off" :: "v"(lb0), "v"(gb0) : "memory");
        if constexpr (BN == 128) {
            const bf16_t* gb1 = Bt + (size_t)(gn0 + brB1) * K + k0 + bcB1 * 8;
            unsigned lb1 = sb + LDSA_B + brB1 * 80 + bcB1 * 16;
            asm volatile("global_load_async_to_lds_b128 %0, %1, off" :: "v"(lb1), "v"(gb1) : "memory");
        }
    };

    v8f acc[2][NF];
#pragma unroll
    for (int s = 0; s < 2; s++)
#pragma unroll
        for (int j = 0; j < NF; j++)
#pragma unroll
            for (int r = 0; r < 8; r++) acc[s][j][r] = 0.f;

    const int NK = K >> 5;
    stage(0, 0);
    for (int kc = 0; kc < NK; ++kc) {
        const int cur = kc & 1;
        if (kc + 1 < NK) {
            stage(kc + 1, cur ^ 1);               // overlap next DMA with compute
            if constexpr (BN == 128)
                asm volatile("s_wait_asynccnt 0x4" ::: "memory");
            else
                asm volatile("s_wait_asynccnt 0x3" ::: "memory");
        } else {
            asm volatile("s_wait_asynccnt 0x0" ::: "memory");
        }
        __syncthreads();                          // cur tile visible to all waves

        const char* bufc = smem + cur * BUF_B;
        v16bf av[2];
#pragma unroll
        for (int s = 0; s < 2; s++) {
            const char* pa = bufc + (wm * 32 + s * 16 + lm) * 80 + hf * 16;
            v8bf aL = *(const v8bf*)pa;
            v8bf aH = *(const v8bf*)(pa + 32);
#pragma unroll
            for (int i = 0; i < 8; i++) { av[s][i] = aL[i]; av[s][8 + i] = aH[i]; }
        }
#pragma unroll
        for (int j = 0; j < NF; j++) {
            const char* pb = bufc + LDSA_B + (wn * (BN / 2) + j * 16 + lm) * 80 + hf * 16;
            v8bf bL = *(const v8bf*)pb;
            v8bf bH = *(const v8bf*)(pb + 32);
            v16bf bv;
#pragma unroll
            for (int i = 0; i < 8; i++) { bv[i] = bL[i]; bv[8 + i] = bH[i]; }
#pragma unroll
            for (int s = 0; s < 2; s++)
                acc[s][j] = __builtin_amdgcn_wmma_f32_16x16x32_bf16(
                    false, av[s], false, bv, (short)0, acc[s][j], false, false);
        }
        __syncthreads();                          // done reading cur before restage
    }

#pragma unroll
    for (int s = 0; s < 2; s++)
#pragma unroll
        for (int r = 0; r < 8; r++) {
            int rowi = gm0 + wm * 32 + s * 16 + r + hf * 8;   // VGPR r -> M=r (+8 hi half)
#pragma unroll
            for (int j = 0; j < NF; j++) {
                int col = gn0 + wn * (BN / 2) + j * 16 + lm;
                float v = acc[s][j][r];
                if (BIAS) v += bias[col];
                if (ACT == 1) v = fmaxf(v, 0.f);
                if (ACT == 2) v = (v > 20.f) ? v : log1pf(__expf(v));   // softplus
                if (RES) v += res[(size_t)rowi * Nn + col];
                if (Cf) Cf[(size_t)rowi * Nn + col] = v;
                if (Cb) Cb[(size_t)rowi * Nn + col] = (bf16_t)v;
            }
        }
}

// ---------------- selective scan (sequential over L) ----------------
__global__ void k_scan(const float* __restrict__ delta, const float* __restrict__ xc,
                       const float* __restrict__ dbc, const float* __restrict__ xz,
                       const float* __restrict__ A_log, const float* __restrict__ Dp,
                       bf16_t* __restrict__ yb) {
    int idx = blockIdx.x * blockDim.x + threadIdx.x;
    if (idx >= BATCH * EDIM) return;
    int e = idx % EDIM, b = idx / EDIM;
    float a[NST], st[NST];
#pragma unroll
    for (int n = 0; n < NST; n++) { a[n] = -__expf(A_log[e * NST + n]); st[n] = 0.f; }
    float Dv = Dp[e];
    for (int t = 0; t < SEQ; t++) {
        size_t row = (size_t)b * SEQ + t;
        if (t + 1 < SEQ) {   // prefetch next timestep activations
            __builtin_prefetch(delta + (row + 1) * EDIM + e, 0, 1);
            __builtin_prefetch(xc    + (row + 1) * EDIM + e, 0, 1);
        }
        float dl = delta[row * EDIM + e];
        float xv = xc[row * EDIM + e];
        float zz = xz[row * (2 * EDIM) + EDIM + e];
        const float* bc = dbc + row * (DTR + 2 * NST);   // [0,32)=dt-in, [32,48)=B, [48,64)=C
        float du = dl * xv;
        float y = 0.f;
#pragma unroll
        for (int n = 0; n < NST; n++) {
            float dA = __expf(dl * a[n]);
            st[n] = dA * st[n] + du * bc[DTR + n];
            y += st[n] * bc[DTR + NST + n];
        }
        y += Dv * xv;
        float sz = zz / (1.f + __expf(-zz));             // SiLU(z)
        yb[row * EDIM + e] = (bf16_t)(y * sz);
    }
}

__global__ void k_add2(const float* __restrict__ a, const float* __restrict__ b,
                       float* __restrict__ o, int n) {
    int i = blockIdx.x * blockDim.x + threadIdx.x;
    if (i < n) o[i] = a[i] + b[i];
}

// ---------------- host orchestration ----------------
static inline size_t alignup(size_t x) { return (x + 255) & ~(size_t)255; }

extern "C" void kernel_launch(void* const* d_in, const int* in_sizes, int n_in,
                              void* d_out, int out_size, void* d_ws, size_t ws_size,
                              hipStream_t stream) {
    (void)in_sizes; (void)n_in; (void)out_size; (void)ws_size;
    const float* x = (const float*)d_in[0];
    // dir params: base 1 (fwd) / 11 (bwd): ln_w,in_w,conv_w,conv_b,xp_w,dt_w,dt_b,A_log,D,out_w
    const float* norm_w[2] = { (const float*)d_in[21], (const float*)d_in[22] };
    const float* ffn_w1 = (const float*)d_in[23];
    const float* ffn_b1 = (const float*)d_in[24];
    const float* ffn_w2 = (const float*)d_in[25];
    const float* ffn_b2 = (const float*)d_in[26];

    // ---- workspace bump allocator ----
    char* base = (char*)d_ws; size_t off = 0;
    auto alloc = [&](size_t bytes) { char* p = base + off; off += alignup(bytes); return (void*)p; };

    float*  xrev   = (float*) alloc((size_t)MROWS * DM * 4);
    bf16_t* in_wt[2]; bf16_t* xp_wt[2]; bf16_t* dt_wt[2]; bf16_t* out_wt[2];
    for (int d = 0; d < 2; d++) {
        in_wt[d]  = (bf16_t*)alloc((size_t)2 * EDIM * DM * 2);
        xp_wt[d]  = (bf16_t*)alloc((size_t)(DTR + 2 * NST) * EDIM * 2);
        dt_wt[d]  = (bf16_t*)alloc((size_t)EDIM * DTR * 2);
        out_wt[d] = (bf16_t*)alloc((size_t)DM * EDIM * 2);
    }
    bf16_t* w1t   = (bf16_t*)alloc((size_t)DFF * DM * 2);
    bf16_t* w2t   = (bf16_t*)alloc((size_t)DM * DFF * 2);

    bf16_t* hbf   = (bf16_t*)alloc((size_t)MROWS * DM * 2);
    float*  xz    = (float*) alloc((size_t)MROWS * 2 * EDIM * 4);
    float*  xc    = (float*) alloc((size_t)MROWS * EDIM * 4);
    bf16_t* xcbf  = (bf16_t*)alloc((size_t)MROWS * EDIM * 2);
    float*  dbc   = (float*) alloc((size_t)MROWS * 64 * 4);
    bf16_t* dbcbf = (bf16_t*)alloc((size_t)MROWS * 64 * 2);
    float*  delta = (float*) alloc((size_t)MROWS * EDIM * 4);
    bf16_t* ybf   = (bf16_t*)alloc((size_t)MROWS * EDIM * 2);
    float*  mfp   = (float*) alloc((size_t)MROWS * DM * 4);
    float*  mn    = (float*) alloc((size_t)MROWS * DM * 4);
    bf16_t* mnbf  = (bf16_t*)alloc((size_t)MROWS * DM * 2);
    bf16_t* f1bf  = (bf16_t*)alloc((size_t)MROWS * DFF * 2);
    float*  outd[2];
    outd[0] = (float*)alloc((size_t)MROWS * DM * 4);
    outd[1] = (float*)alloc((size_t)MROWS * DM * 4);

    const int T = 256;
    auto gblk = [](int Mr, int Nn, int BN) { return (Mr / 128) * (Nn / BN); };

    // reversed input for the backward direction
    k_reverse<<<(MROWS * DM + T - 1) / T, T, 0, stream>>>(x, xrev);

    // weight conversion (+ transpose to N-major for the WMMA B operand)
    for (int d = 0; d < 2; d++) {
        int pb = 1 + d * 10;
        k_wt<<<(DM * 2 * EDIM + T - 1) / T, T, 0, stream>>>((const float*)d_in[pb + 1], in_wt[d], DM, 2 * EDIM);
        k_wt<<<(EDIM * 64 + T - 1) / T, T, 0, stream>>>((const float*)d_in[pb + 4], xp_wt[d], EDIM, DTR + 2 * NST);
        k_wt<<<(DTR * EDIM + T - 1) / T, T, 0, stream>>>((const float*)d_in[pb + 5], dt_wt[d], DTR, EDIM);
        k_wt<<<(EDIM * DM + T - 1) / T, T, 0, stream>>>((const float*)d_in[pb + 9], out_wt[d], EDIM, DM);
    }
    k_wt<<<(DM * DFF + T - 1) / T, T, 0, stream>>>(ffn_w1, w1t, DM, DFF);
    k_wt<<<(DFF * DM + T - 1) / T, T, 0, stream>>>(ffn_w2, w2t, DFF, DM);

    for (int d = 0; d < 2; d++) {
        int pb = 1 + d * 10;
        const float* xd     = d ? xrev : x;
        const float* ln_w   = (const float*)d_in[pb + 0];
        const float* conv_w = (const float*)d_in[pb + 2];
        const float* conv_b = (const float*)d_in[pb + 3];
        const float* dt_b   = (const float*)d_in[pb + 6];
        const float* A_log  = (const float*)d_in[pb + 7];
        const float* Dp     = (const float*)d_in[pb + 8];

        // RMSNorm -> bf16
        k_rmsnorm<<<MROWS, T, 0, stream>>>(xd, ln_w, nullptr, hbf, DM);
        // xz = h @ in_w                            (2048x512 x 512x2048)
        k_gemm<128, 0, false, false><<<gblk(MROWS, 2 * EDIM, 128), T, 0, stream>>>(
            hbf, DM, in_wt[d], nullptr, nullptr, xz, nullptr, MROWS, 2 * EDIM, DM);
        // depthwise conv + SiLU
        k_conv<<<(MROWS * EDIM + T - 1) / T, T, 0, stream>>>(xz, conv_w, conv_b, xc, xcbf);
        // dbc = xc @ xp_w                          (2048x1024 x 1024x64)
        k_gemm<64, 0, false, false><<<gblk(MROWS, 64, 64), T, 0, stream>>>(
            xcbf, EDIM, xp_wt[d], nullptr, nullptr, dbc, dbcbf, MROWS, 64, EDIM);
        // delta = softplus(dbc[:, :32] @ dt_w + dt_b)   (K=32, lda=64)
        k_gemm<128, 2, true, false><<<gblk(MROWS, EDIM, 128), T, 0, stream>>>(
            dbcbf, 64, dt_wt[d], dt_b, nullptr, delta, nullptr, MROWS, EDIM, DTR);
        // selective scan + gate with SiLU(z) -> bf16 y
        k_scan<<<(BATCH * EDIM + T - 1) / T, T, 0, stream>>>(delta, xc, dbc, xz, A_log, Dp, ybf);
        // m = y @ out_w + x  (residual)
        k_gemm<128, 0, false, true><<<gblk(MROWS, DM, 128), T, 0, stream>>>(
            ybf, EDIM, out_wt[d], nullptr, xd, mfp, nullptr, MROWS, DM, EDIM);
        // mn = rmsnorm(m) -> fp32 + bf16
        k_rmsnorm<<<MROWS, T, 0, stream>>>(mfp, norm_w[d], mn, mnbf, DM);
        // f1 = relu(mn @ w1 + b1) -> bf16
        k_gemm<128, 1, true, false><<<gblk(MROWS, DFF, 128), T, 0, stream>>>(
            mnbf, DM, w1t, ffn_b1, nullptr, nullptr, f1bf, MROWS, DFF, DM);
        // out_d = f1 @ w2 + b2 + mn
        k_gemm<128, 0, true, true><<<gblk(MROWS, DM, 128), T, 0, stream>>>(
            f1bf, DFF, w2t, ffn_b2, mn, outd[d], nullptr, MROWS, DM, DFF);
    }

    k_add2<<<(MROWS * DM + T - 1) / T, T, 0, stream>>>(outd[0], outd[1], (float*)d_out, MROWS * DM);
}
